// PointNetEncoder_28501402976299
// MI455X (gfx1250) — matrix operand order repaired and checked by reading, not compiled, not run on hardware
//
#include <hip/hip_runtime.h>

typedef __bf16 bf16;
typedef bf16 v16bf __attribute__((ext_vector_type(16)));
typedef float v8f __attribute__((ext_vector_type(8)));

#define LN_EPS 1e-5f

// ---------------- LDS layout for the main kernel (bytes) ----------------
#define WGT_ELEMS 40960            // 80 pre-shuffled bf16 fragments * 512
#define WGT_BYTES 81920
#define PRM_OFF   81920            // f32 params
#define XS_OFF    86528            // 8 waves * 32 rows * 4 f32
#define BID_OFF   90624            // 8 waves * 32 int
#define ACT_OFF   91648            // 8 waves * 2 * 2048 bf16 A-fragment tiles
#define SMEM_MAIN 157184

// fragment-element offsets of each weight matrix inside the wgt region
#define FW02 0          // K=64 ,N=128: KC=2, NT=8 -> 16 frags
#define FW11 8192       // K=128,N=64 : KC=4, NT=4 -> 16 frags
#define FW12 16384
#define FW21 24576
#define FW22 32768

// param (float) offsets inside prm[]
#define PW01 0
#define PB01 192
#define PG0  256
#define PBE0 320
#define PB11 384        // folded: b11 + b02@W11
#define PG1  448
#define PBE1 512
#define PB21 576        // folded: b21 + b12@W21
#define PG2  640
#define PBE2 704
#define PB22 768        // 128
// total 896 floats

// ---------------- WMMA helpers ----------------
__device__ __forceinline__ v8f wmma_bf16(v16bf a, v16bf b, v8f c) {
  return __builtin_amdgcn_wmma_f32_16x16x32_bf16(false, a, false, b, (short)0, c,
                                                 false, false);
}
__device__ __forceinline__ v8f v8f_zero() {
  v8f z = {0.f, 0.f, 0.f, 0.f, 0.f, 0.f, 0.f, 0.f};
  return z;
}

// Per-lane base constant of the C/D->A-fragment scatter:
// addr(t,v) = C0 + 512*(t>>1) + 8*(t&1) + 16*v   (elements)
__device__ __forceinline__ int afrag_base(int lane) {
  const int half = lane >> 4, nlo = lane & 15;
  return 128 * half + 256 * ((nlo >> 3) & 1) + (nlo & 7);
}

// LayerNorm(64)+ReLU on a 16x64 tile (4 accumulators); write bf16 in
// A-fragment layout (kc chunks 0..1).
__device__ __forceinline__ void ln_relu_storeA(const v8f (&c)[4],
                                               const float* __restrict__ gam,
                                               const float* __restrict__ bet,
                                               bf16* __restrict__ act, int lane) {
  const int nlo = lane & 15;
  bf16* base = act + afrag_base(lane);
#pragma unroll
  for (int v = 0; v < 8; ++v) {
    float p = c[0][v] + c[1][v] + c[2][v] + c[3][v];
    float q = c[0][v] * c[0][v] + c[1][v] * c[1][v] + c[2][v] * c[2][v] +
              c[3][v] * c[3][v];
#pragma unroll
    for (int m = 1; m < 16; m <<= 1) {  // reduce within 16-lane half (wave32)
      p += __shfl_xor(p, m, 32);
      q += __shfl_xor(q, m, 32);
    }
    float mean = p * (1.f / 64.f);
    float var = q * (1.f / 64.f) - mean * mean;
    float rstd = rsqrtf(var + LN_EPS);
#pragma unroll
    for (int t = 0; t < 4; ++t) {
      int col = nlo + (t << 4);
      float val = (c[t][v] - mean) * rstd * gam[col] + bet[col];
      base[512 * (t >> 1) + 8 * (t & 1) + 16 * v] = (bf16)fmaxf(val, 0.f);
    }
  }
}

// GEMM (KC=2, NT=8): act64 @ W[64,128], M=32 (two A tiles) so every
// B fragment feeds 2 WMMAs. Bias is algebraically folded into the *next*
// stage, so the store is a pure convert. 2 live accumulators.
__device__ __forceinline__ void gemm_storeA(const bf16* __restrict__ Bw,
                                            bf16* __restrict__ act0,
                                            bf16* __restrict__ act1, int lane) {
  const bf16* bp = Bw + lane * 16;
  v16bf a[2][2];
  a[0][0] = *(const v16bf*)(act0 + lane * 16);
  a[0][1] = *(const v16bf*)(act0 + 512 + lane * 16);
  a[1][0] = *(const v16bf*)(act1 + lane * 16);
  a[1][1] = *(const v16bf*)(act1 + 512 + lane * 16);
  bf16* s0 = act0 + afrag_base(lane);
  bf16* s1 = act1 + afrag_base(lane);
#pragma unroll
  for (int nt = 0; nt < 8; ++nt) {
    v16bf B0 = *(const v16bf*)(bp + (nt * 2 + 0) * 512);
    v16bf B1 = *(const v16bf*)(bp + (nt * 2 + 1) * 512);
    v8f c0 = wmma_bf16(a[0][0], B0, v8f_zero());
    c0 = wmma_bf16(a[0][1], B1, c0);
    v8f c1 = wmma_bf16(a[1][0], B0, v8f_zero());
    c1 = wmma_bf16(a[1][1], B1, c1);
    int off = 512 * (nt >> 1) + 8 * (nt & 1);
#pragma unroll
    for (int v = 0; v < 8; ++v) {
      s0[off + 16 * v] = (bf16)c0[v];
      s1[off + 16 * v] = (bf16)c1[v];
    }
  }
}

// GEMM (KC=4, NT=4): act128 @ W[128,64] + folded bias -> LN+ReLU -> act64.
// M=32: B fragments reused by both M tiles.
__device__ __forceinline__ void gemm_lnA(const bf16* __restrict__ Bw,
                                         const float* __restrict__ b1,
                                         const float* __restrict__ gam,
                                         const float* __restrict__ bet,
                                         bf16* __restrict__ act0,
                                         bf16* __restrict__ act1, int lane) {
  const bf16* bp = Bw + lane * 16;
  const int nlo = lane & 15;
  v16bf a[2][4];
#pragma unroll
  for (int kc = 0; kc < 4; ++kc) {
    a[0][kc] = *(const v16bf*)(act0 + kc * 512 + lane * 16);
    a[1][kc] = *(const v16bf*)(act1 + kc * 512 + lane * 16);
  }
  v8f c[2][4];
#pragma unroll
  for (int nt = 0; nt < 4; ++nt) {
    v16bf B0 = *(const v16bf*)(bp + (nt * 4 + 0) * 512);
    v16bf B1 = *(const v16bf*)(bp + (nt * 4 + 1) * 512);
    v16bf B2 = *(const v16bf*)(bp + (nt * 4 + 2) * 512);
    v16bf B3 = *(const v16bf*)(bp + (nt * 4 + 3) * 512);
#pragma unroll
    for (int mt = 0; mt < 2; ++mt) {
      v8f cc = wmma_bf16(a[mt][0], B0, v8f_zero());
      cc = wmma_bf16(a[mt][1], B1, cc);
      cc = wmma_bf16(a[mt][2], B2, cc);
      cc = wmma_bf16(a[mt][3], B3, cc);
      c[mt][nt] = cc;
    }
  }
#pragma unroll
  for (int nt = 0; nt < 4; ++nt) {
    float bb = b1[nlo + (nt << 4)];
#pragma unroll
    for (int mt = 0; mt < 2; ++mt)
#pragma unroll
      for (int v = 0; v < 8; ++v) c[mt][nt][v] += bb;
  }
  ln_relu_storeA(c[0], gam, bet, act0, lane);
  ln_relu_storeA(c[1], gam, bet, act1, lane);
}

// Final GEMM (KC=2, NT=8): act64 @ W[64,128] + bias, fused 16-row max
// reduction + global atomic f32 max into the per-cloud feature rows.
__device__ __forceinline__ void gemm_maxout(const bf16* __restrict__ Bw,
                                            const float* __restrict__ b22,
                                            const bf16* __restrict__ act0,
                                            const bf16* __restrict__ act1,
                                            const int* __restrict__ bids,
                                            float* __restrict__ seg, int lane) {
  const bf16* bp = Bw + lane * 16;
  const int nlo = lane & 15, half = lane >> 4;
  v16bf a[2][2];
  a[0][0] = *(const v16bf*)(act0 + lane * 16);
  a[0][1] = *(const v16bf*)(act0 + 512 + lane * 16);
  a[1][0] = *(const v16bf*)(act1 + lane * 16);
  a[1][1] = *(const v16bf*)(act1 + 512 + lane * 16);
  const int bid0 = bids[0], bid1 = bids[16];
  const bool u0 = (bid0 == bids[15]) && (bid0 >= 0);  // invalid rows have -1
  const bool u1 = (bid1 == bids[31]) && (bid1 >= 0);
#pragma unroll
  for (int nt = 0; nt < 8; ++nt) {
    v16bf B0 = *(const v16bf*)(bp + (nt * 2 + 0) * 512);
    v16bf B1 = *(const v16bf*)(bp + (nt * 2 + 1) * 512);
    float bb = b22[nlo + (nt << 4)];
    int col = nlo + (nt << 4);
    v8f c0 = wmma_bf16(a[0][0], B0, v8f_zero());
    c0 = wmma_bf16(a[0][1], B1, c0);
    v8f c1 = wmma_bf16(a[1][0], B0, v8f_zero());
    c1 = wmma_bf16(a[1][1], B1, c1);
#pragma unroll
    for (int mt = 0; mt < 2; ++mt) {
      const v8f& cc = mt ? c1 : c0;
      bool uni = mt ? u1 : u0;
      int bid = mt ? bid1 : bid0;
      if (uni) {
        float m8 = cc[0];
#pragma unroll
        for (int v = 1; v < 8; ++v) m8 = fmaxf(m8, cc[v]);
        float m16 = fmaxf(m8, __shfl_xor(m8, 16, 32));
        if (half == (nt & 1))  // one lane-half issues the 16 column atomics
          __hip_atomic_fetch_max(&seg[bid * 128 + col], m16 + bb,
                                 __ATOMIC_RELAXED, __HIP_MEMORY_SCOPE_AGENT);
      } else {
#pragma unroll
        for (int v = 0; v < 8; ++v) {
          int row = mt * 16 + v + (half << 3);
          int b = bids[row];
          if (b >= 0)
            __hip_atomic_fetch_max(&seg[b * 128 + col], cc[v] + bb,
                                   __ATOMIC_RELAXED, __HIP_MEMORY_SCOPE_AGENT);
        }
      }
    }
  }
}

// ---------------- helper kernels ----------------
__global__ void pn_init_seg(float* __restrict__ seg) {
  int i = blockIdx.x * 256 + threadIdx.x;
  if (i < 64 * 128) seg[i] = -3.402823466e38f;
}

// Convert weights to bf16 pre-shuffled into WMMA B-fragment lane order:
// dst[matBase + (nt*KC+kc)*512 + lane*16 + e] =
//   W[(kc*32 + 8*(lane>=16) + (e&7) + 16*(e>=8)) * ldn + nt*16 + (lane&15)]
__global__ void pn_convert_weights(const float* __restrict__ W02,
                                   const float* __restrict__ W11,
                                   const float* __restrict__ W12,
                                   const float* __restrict__ W21,
                                   const float* __restrict__ W22,
                                   bf16* __restrict__ dst) {
  int i = blockIdx.x * 256 + threadIdx.x;
  if (i >= WGT_ELEMS) return;
  int mat = i >> 13;       // 8192 elements per matrix
  int within = i & 8191;
  int f = within >> 9;     // fragment (512 elements)
  int r = within & 511;
  int lane = r >> 4;
  int e = r & 15;
  const float* W;
  int ldn, KC;
  switch (mat) {
    case 0:  W = W02; ldn = 128; KC = 2; break;
    case 1:  W = W11; ldn = 64;  KC = 4; break;
    case 2:  W = W12; ldn = 128; KC = 2; break;
    case 3:  W = W21; ldn = 64;  KC = 4; break;
    default: W = W22; ldn = 128; KC = 2; break;
  }
  int nt = f / KC, kc = f % KC;
  int n = (lane & 15) + nt * 16;
  int k = kc * 32 + ((lane & 16) ? 8 : 0) + (e & 7) + ((e & 8) ? 16 : 0);
  dst[i] = (bf16)W[k * ldn + n];
}

// Fold the 128-wide biases through the following 128->64 weights:
// b11f = b11 + b02@W11 ; b21f = b21 + b12@W21   (exact f32)
__global__ void pn_fold_bias(const float* __restrict__ b02,
                             const float* __restrict__ W11,
                             const float* __restrict__ b11,
                             const float* __restrict__ b12,
                             const float* __restrict__ W21,
                             const float* __restrict__ b21,
                             float* __restrict__ b11f,
                             float* __restrict__ b21f) {
  int j = threadIdx.x;  // 64 threads
  float s1 = b11[j], s2 = b21[j];
  for (int k = 0; k < 128; ++k) {
    s1 += b02[k] * W11[k * 64 + j];
    s2 += b12[k] * W21[k * 64 + j];
  }
  b11f[j] = s1;
  b21f[j] = s2;
}

__global__ void pn_counts(const int* __restrict__ batch, int n,
                          int* __restrict__ counts, int* __restrict__ haspad) {
  __shared__ int sc[64];
  __shared__ int mx;
  int b = threadIdx.x;  // 64 threads; batch is sorted -> binary search bounds
  int lo = 0, hi = n;
  while (lo < hi) { int mid = (lo + hi) >> 1; if (batch[mid] <  b) lo = mid + 1; else hi = mid; }
  int first = lo;
  lo = 0; hi = n;
  while (lo < hi) { int mid = (lo + hi) >> 1; if (batch[mid] <= b) lo = mid + 1; else hi = mid; }
  int cnt = lo - first;
  sc[b] = cnt;
  __syncthreads();
  if (b == 0) { int m = 0; for (int i = 0; i < 64; ++i) m = m > sc[i] ? m : sc[i]; mx = m; }
  __syncthreads();
  counts[b] = cnt;
  haspad[b] = (cnt < mx) ? 1 : 0;
}

// one point-MLP layer in exact f32 (for the zero-point padding feature)
__device__ void v0_layer(const float* __restrict__ cur, int indim,
                         const float* __restrict__ W1, const float* __restrict__ b1,
                         const float* __restrict__ gam, const float* __restrict__ bet,
                         const float* __restrict__ W2, const float* __restrict__ b2,
                         float* __restrict__ out, float* __restrict__ hbuf,
                         float* __restrict__ mv) {
  int tid = threadIdx.x;  // blockDim == 128
  if (tid < 64) {
    float h = b1[tid];
    for (int k = 0; k < indim; ++k) h += cur[k] * W1[k * 64 + tid];
    hbuf[tid] = h;
  }
  __syncthreads();
  if (tid == 0) {
    float s = 0.f, q = 0.f;
    for (int i = 0; i < 64; ++i) { s += hbuf[i]; q += hbuf[i] * hbuf[i]; }
    float mean = s * (1.f / 64.f);
    mv[0] = mean;
    mv[1] = rsqrtf(q * (1.f / 64.f) - mean * mean + LN_EPS);
  }
  __syncthreads();
  if (tid < 64) hbuf[tid] = fmaxf((hbuf[tid] - mv[0]) * mv[1] * gam[tid] + bet[tid], 0.f);
  __syncthreads();
  {
    float o = b2[tid];
    for (int k = 0; k < 64; ++k) o += hbuf[k] * W2[k * 128 + tid];
    out[tid] = o;
  }
  __syncthreads();
}

__global__ void pn_v0(const float* W10, const float* b10, const float* g0, const float* be0,
                      const float* W20, const float* b20,
                      const float* W11, const float* b11, const float* g1, const float* be1,
                      const float* W21, const float* b21,
                      const float* W12, const float* b12, const float* g2, const float* be2,
                      const float* W22, const float* b22, float* __restrict__ v0out) {
  __shared__ float bufA[128], bufB[128], hbuf[64], mv[2];
  int tid = threadIdx.x;
  if (tid < 3) bufA[tid] = 0.f;
  __syncthreads();
  v0_layer(bufA, 3,   W10, b10, g0, be0, W20, b20, bufB, hbuf, mv);
  v0_layer(bufB, 128, W11, b11, g1, be1, W21, b21, bufA, hbuf, mv);
  v0_layer(bufA, 128, W12, b12, g2, be2, W22, b22, bufB, hbuf, mv);
  v0out[tid] = bufB[tid];
}

// ---------------- main fused per-point kernel (M=32 per wave) ----------------
__global__ __launch_bounds__(256) void pn_main(
    const float* __restrict__ x, const int* __restrict__ batch, int npts,
    const float* __restrict__ W01, const float* __restrict__ b01,
    const float* __restrict__ g0, const float* __restrict__ be0,
    const float* __restrict__ b11f, const float* __restrict__ g1,
    const float* __restrict__ be1, const float* __restrict__ b21f,
    const float* __restrict__ g2, const float* __restrict__ be2,
    const float* __restrict__ b22, const bf16* __restrict__ wbf,
    float* __restrict__ seg) {
  extern __shared__ char smem[];
  bf16* wgt = (bf16*)smem;
  float* prm = (float*)(smem + PRM_OFF);
  float* xs_all = (float*)(smem + XS_OFF);
  int* bid_all = (int*)(smem + BID_OFF);
  bf16* act_all = (bf16*)(smem + ACT_OFF);

  const int tid = threadIdx.x;
  // stage pre-shuffled bf16 weight fragments (81920B) as int4
  {
    int4* d = (int4*)wgt;
    const int4* s = (const int4*)wbf;
    for (int i = tid; i < WGT_BYTES / 16; i += 256) d[i] = s[i];
  }
  // stage f32 params
  if (tid < 192) prm[PW01 + tid] = W01[tid];
  if (tid < 64)  prm[PB01 + tid] = b01[tid];
  if (tid < 64)  prm[PG0  + tid] = g0[tid];
  if (tid < 64)  prm[PBE0 + tid] = be0[tid];
  if (tid < 64)  prm[PB11 + tid] = b11f[tid];
  if (tid < 64)  prm[PG1  + tid] = g1[tid];
  if (tid < 64)  prm[PBE1 + tid] = be1[tid];
  if (tid < 64)  prm[PB21 + tid] = b21f[tid];
  if (tid < 64)  prm[PG2  + tid] = g2[tid];
  if (tid < 64)  prm[PBE2 + tid] = be2[tid];
  if (tid < 128) prm[PB22 + tid] = b22[tid];
  __syncthreads();

  const int wave = tid >> 5, lane = tid & 31;
  const int half = lane >> 4, nlo = lane & 15;
  float* xs = xs_all + wave * 128;   // 32 rows * 4
  int* bids = bid_all + wave * 32;
  bf16* act0 = act_all + wave * 4096;
  bf16* act1 = act0 + 2048;

  const int groups = (npts + 31) >> 5;  // 32 points per wave-iteration
  const int gstride = gridDim.x * 8;
  for (int grp = blockIdx.x * 8 + wave; grp < groups; grp += gstride) {
    const int base = grp << 5;
    // prefetch next group's points (global_prefetch_b8)
    int nxt = grp + gstride;
    if (nxt < groups) __builtin_prefetch(x + (size_t)(nxt << 5) * 3, 0, 0);
    // stage 32 x-rows + batch ids (one row per lane)
    {
      int idx = base + lane;
      bool valid = idx < npts;
      int s = valid ? idx : 0;
      xs[lane * 4 + 0] = valid ? x[s * 3 + 0] : 0.f;
      xs[lane * 4 + 1] = valid ? x[s * 3 + 1] : 0.f;
      xs[lane * 4 + 2] = valid ? x[s * 3 + 2] : 0.f;
      bids[lane] = valid ? batch[s] : -1;
    }

    // --- layer0 first linear (3->64) on VALU + LN + ReLU, per M tile ---
#pragma unroll
    for (int mt = 0; mt < 2; ++mt) {
      v8f c4[4];
#pragma unroll
      for (int t = 0; t < 4; ++t) {
#pragma unroll
        for (int v = 0; v < 8; ++v) {
          int row = mt * 16 + v + (half << 3);
          int col = nlo + (t << 4);
          float h = prm[PB01 + col];
          h += xs[row * 4 + 0] * prm[PW01 + 0 * 64 + col];
          h += xs[row * 4 + 1] * prm[PW01 + 1 * 64 + col];
          h += xs[row * 4 + 2] * prm[PW01 + 2 * 64 + col];
          c4[t][v] = h;
        }
      }
      ln_relu_storeA(c4, prm + PG0, prm + PBE0, mt ? act1 : act0, lane);
    }

    // layer0: act64 @ W2[64,128] -> act128 (bias folded into b11f)
    gemm_storeA(wgt + FW02, act0, act1, lane);
    // layer1: act128 @ W1[128,64] (+b11f) -> LN+ReLU -> act64
    gemm_lnA(wgt + FW11, prm + PB11, prm + PG1, prm + PBE1, act0, act1, lane);
    // layer1: act64 @ W2[64,128] -> act128 (bias folded into b21f)
    gemm_storeA(wgt + FW12, act0, act1, lane);
    // layer2: act128 @ W1[128,64] (+b21f) -> LN+ReLU -> act64
    gemm_lnA(wgt + FW21, prm + PB21, prm + PG2, prm + PBE2, act0, act1, lane);
    // layer2: act64 @ W2[64,128] (+b22) -> fused segment max + atomics
    gemm_maxout(wgt + FW22, prm + PB22, act0, act1, bids, seg, lane);
  }
}

// ---------------- final global transform (exact f32, tiny) ----------------
__global__ void pn_final(const float* __restrict__ seg, const int* __restrict__ haspad,
                         const float* __restrict__ v0, const float* __restrict__ W1,
                         const float* __restrict__ b1, const float* __restrict__ gam,
                         const float* __restrict__ bet, const float* __restrict__ W2,
                         const float* __restrict__ b2, float* __restrict__ out) {
  int b = threadIdx.x;  // 64 threads, one cloud each
  float gv[128];
  bool hp = haspad[b] != 0;
  for (int j = 0; j < 128; ++j) {
    float s = seg[b * 128 + j];
    if (hp) s = fmaxf(s, v0[j]);
    gv[j] = s;
  }
  float h[128];
  float sum = 0.f, sq = 0.f;
  for (int j = 0; j < 128; ++j) {
    float a = b1[j];
    for (int k = 0; k < 128; ++k) a += gv[k] * W1[k * 128 + j];
    h[j] = a; sum += a; sq += a * a;
  }
  float mean = sum * (1.f / 128.f);
  float rstd = rsqrtf(sq * (1.f / 128.f) - mean * mean + LN_EPS);
  for (int j = 0; j < 128; ++j)
    h[j] = fmaxf((h[j] - mean) * rstd * gam[j] + bet[j], 0.f);
  for (int j = 0; j < 128; ++j) {
    float o = b2[j];
    for (int k = 0; k < 128; ++k) o += h[k] * W2[k * 128 + j];
    out[b * 128 + j] = o;
  }
}

// ---------------- launch ----------------
extern "C" void kernel_launch(void* const* d_in, const int* in_sizes, int n_in,
                              void* d_out, int out_size, void* d_ws, size_t ws_size,
                              hipStream_t stream) {
  // setup_inputs order, tuples flattened:
  // 0:x  1-6:layer0(W1,b1,g,be,W2,b2)  7-12:layer1  13-18:layer2  19-24:gt  25:batch
  const float* x    = (const float*)d_in[0];
  const float* l0W1 = (const float*)d_in[1];
  const float* l0b1 = (const float*)d_in[2];
  const float* l0g  = (const float*)d_in[3];
  const float* l0be = (const float*)d_in[4];
  const float* l0W2 = (const float*)d_in[5];
  const float* l0b2 = (const float*)d_in[6];
  const float* l1W1 = (const float*)d_in[7];
  const float* l1b1 = (const float*)d_in[8];
  const float* l1g  = (const float*)d_in[9];
  const float* l1be = (const float*)d_in[10];
  const float* l1W2 = (const float*)d_in[11];
  const float* l1b2 = (const float*)d_in[12];
  const float* l2W1 = (const float*)d_in[13];
  const float* l2b1 = (const float*)d_in[14];
  const float* l2g  = (const float*)d_in[15];
  const float* l2be = (const float*)d_in[16];
  const float* l2W2 = (const float*)d_in[17];
  const float* l2b2 = (const float*)d_in[18];
  const float* gtW1 = (const float*)d_in[19];
  const float* gtb1 = (const float*)d_in[20];
  const float* gtg  = (const float*)d_in[21];
  const float* gtbe = (const float*)d_in[22];
  const float* gtW2 = (const float*)d_in[23];
  const float* gtb2 = (const float*)d_in[24];
  const int* batch  = (const int*)d_in[25];
  const int npts = in_sizes[0] / 3;

  char* ws = (char*)d_ws;
  bf16* wbf    = (bf16*)ws;                  // 40960 bf16  (81920 B)
  float* seg   = (float*)(ws + 81920);       // 64*128 f32  (32768 B)
  float* v0    = (float*)(ws + 114688);      // 128 f32
  int* counts  = (int*)(ws + 115200);        // 64 int
  int* haspad  = (int*)(ws + 115456);        // 64 int
  float* b11f  = (float*)(ws + 115712);      // 64 f32 (folded bias)
  float* b21f  = (float*)(ws + 115968);      // 64 f32 (folded bias)
  float* outp  = (float*)d_out;

  pn_init_seg<<<32, 256, 0, stream>>>(seg);
  pn_convert_weights<<<(WGT_ELEMS + 255) / 256, 256, 0, stream>>>(
      l0W2, l1W1, l1W2, l2W1, l2W2, wbf);
  pn_fold_bias<<<1, 64, 0, stream>>>(l0b2, l1W1, l1b1, l1b2, l2W1, l2b1,
                                     b11f, b21f);
  pn_counts<<<1, 64, 0, stream>>>(batch, npts, counts, haspad);
  pn_v0<<<1, 128, 0, stream>>>(l0W1, l0b1, l0g, l0be, l0W2, l0b2,
                               l1W1, l1b1, l1g, l1be, l1W2, l1b2,
                               l2W1, l2b1, l2g, l2be, l2W2, l2b2, v0);

  int groups = (npts + 31) / 32;
  int grid = (groups + 7) / 8;
  if (grid > 768) grid = 768;
  if (grid < 1) grid = 1;
  pn_main<<<grid, 256, SMEM_MAIN, stream>>>(
      x, batch, npts, l0W1, l0b1, l0g, l0be, b11f, l1g, l1be, b21f,
      l2g, l2be, l2b2, wbf, seg);

  pn_final<<<1, 64, 0, stream>>>(seg, haspad, v0, gtW1, gtb1, gtg, gtbe, gtW2,
                                 gtb2, outp);
}
